// ExpertParallelLayer_83116207112405
// MI455X (gfx1250) — matrix operand order repaired and checked by reading, not compiled, not run on hardware
//
#include <hip/hip_runtime.h>
#include <math.h>

typedef __attribute__((ext_vector_type(8)))  float  v8f;
typedef __attribute__((ext_vector_type(16))) __bf16 v16bf;
typedef __attribute__((ext_vector_type(8)))  __bf16 v8bf;

#define T_TOK 8192
#define DIM   1024
#define HID   2048
#define NEXP  8
#define MT    32   // tokens per tile (2 fragment row-halves)

// workspace layout (bytes); total ~64.3 MB
#define OFF_EXPERT 0
#define OFF_PROB   (T_TOK * 4)
#define OFF_COUNTS (2 * T_TOK * 4)
#define OFF_PERM   (OFF_COUNTS + 1024)
#define OFF_W1T    ((size_t)(OFF_PERM + NEXP * T_TOK * 4))
#define OFF_W2T    (OFF_W1T + (size_t)NEXP * HID * DIM * 2)

#define XSS (DIM + 8)   // xs row stride (bf16 elems): 2064B % 256 == 16
#define HSS (HID + 8)   // hs row stride (bf16 elems): 4112B % 256 == 16

// A fragment (16-bit 16x32): lane m=L&15, kh=L>>4 holds contiguous K run
// [kk+16kh, +16)
static __device__ __forceinline__ v16bf ldA(const __bf16* row, int kk, int kh)
{
    v8bf a0 = *(const v8bf*)(row + kk + 16 * kh);
    v8bf a1 = *(const v8bf*)(row + kk + 16 * kh + 8);
    return __builtin_shufflevector(a0, a1,
        0,1,2,3,4,5,6,7,8,9,10,11,12,13,14,15);
}
// B fragment: lane n=L&15 holds contiguous K run [kk+16kh, +16)
// (col pointer already includes +16*kh)
static __device__ __forceinline__ v16bf ldB(const __bf16* col, int kk)
{
    v8bf b0 = *(const v8bf*)(col + kk);
    v8bf b1 = *(const v8bf*)(col + kk + 8);
    return __builtin_shufflevector(b0, b1,
        0,1,2,3,4,5,6,7,8,9,10,11,12,13,14,15);
}
static __device__ __forceinline__ v8f wmma_bf16(v16bf a, v16bf b, v8f c)
{
    return __builtin_amdgcn_wmma_f32_16x16x32_bf16(
        false, a, false, b, (short)0, c, false, false);
}

// -------------------------------------------------------------------------
// Kernel 1: router. One 64-thread block per token.
// -------------------------------------------------------------------------
__global__ __launch_bounds__(64) void k_router(
    const float* __restrict__ x, const float* __restrict__ Wr,
    const float* __restrict__ br, int* __restrict__ expert,
    float* __restrict__ prob, int* __restrict__ counts)
{
    __shared__ float red[64 * NEXP];
    const int t = blockIdx.x;
    const int tid = threadIdx.x;
    if (t == 0 && tid < NEXP) counts[tid] = 0;

    float acc[NEXP];
#pragma unroll
    for (int e = 0; e < NEXP; ++e) acc[e] = 0.f;

    const float* xr = x + (size_t)t * DIM;
    for (int d = tid; d < DIM; d += 64) {
        float xv = xr[d];
        const float4* w = (const float4*)(Wr + d * NEXP);
        float4 w0 = w[0], w1 = w[1];
        acc[0] += xv * w0.x; acc[1] += xv * w0.y;
        acc[2] += xv * w0.z; acc[3] += xv * w0.w;
        acc[4] += xv * w1.x; acc[5] += xv * w1.y;
        acc[6] += xv * w1.z; acc[7] += xv * w1.w;
    }
#pragma unroll
    for (int e = 0; e < NEXP; ++e) red[tid * NEXP + e] = acc[e];
    __syncthreads();
    for (int s = 32; s > 0; s >>= 1) {
        if (tid < s) {
#pragma unroll
            for (int e = 0; e < NEXP; ++e)
                red[tid * NEXP + e] += red[(tid + s) * NEXP + e];
        }
        __syncthreads();
    }
    if (tid == 0) {
        float l[NEXP];
        float m = -1e30f; int bi = 0;
#pragma unroll
        for (int e = 0; e < NEXP; ++e) l[e] = red[e] + br[e];
#pragma unroll
        for (int e = 0; e < NEXP; ++e) { if (l[e] > m) { m = l[e]; bi = e; } }
        float s = 0.f;
#pragma unroll
        for (int e = 0; e < NEXP; ++e) s += expf(l[e] - m);
        expert[t] = bi;
        prob[t] = 1.0f / s;
    }
}

// -------------------------------------------------------------------------
// Kernel 2: group tokens by expert
// -------------------------------------------------------------------------
__global__ __launch_bounds__(256) void k_scatter(
    const int* __restrict__ expert, int* counts, int* __restrict__ perm)
{
    int t = blockIdx.x * blockDim.x + threadIdx.x;
    int e = expert[t];
    int pos = atomicAdd(&counts[e], 1);
    perm[e * T_TOK + pos] = t;
}

// -------------------------------------------------------------------------
// Kernel 3: transpose + fp32->bf16 weights. in [E][R][C] f32 -> out [E][C][R]
// -------------------------------------------------------------------------
__global__ __launch_bounds__(256) void k_tcvt(
    const float* __restrict__ in, __bf16* __restrict__ out, int R, int C)
{
    __shared__ __bf16 tile[32][33];
    const size_t eoff = (size_t)blockIdx.z * R * C;
    const float* src = in + eoff;
    __bf16* dst = out + eoff;
    const int c0 = blockIdx.x * 32, r0 = blockIdx.y * 32;
    const int tx = threadIdx.x & 31, ty = threadIdx.x >> 5;
#pragma unroll
    for (int i = 0; i < 32; i += 8)
        tile[ty + i][tx] = (__bf16)src[(size_t)(r0 + ty + i) * C + c0 + tx];
    __syncthreads();
#pragma unroll
    for (int i = 0; i < 32; i += 8)
        dst[(size_t)(c0 + ty + i) * R + r0 + tx] = tile[tx][ty + i];
}

// -------------------------------------------------------------------------
// Kernel 4: grouped expert MLP. 32 tokens/block, 512 threads (16 waves).
// Each wave owns 2 n-tiles and computes BOTH 16-row halves, so each B
// fragment feeds 2 WMMAs. 1-step software prefetch (2-stage rotation
// matches unroll 2 -> pure register renaming, no moves, no hazard NOPs).
// -------------------------------------------------------------------------
__global__ __launch_bounds__(512) void k_moe(
    const float* __restrict__ x,
    const __bf16* __restrict__ W1t, const float* __restrict__ b1,
    const __bf16* __restrict__ W2t, const float* __restrict__ b2,
    const int* __restrict__ perm, const int* __restrict__ counts,
    const float* __restrict__ prob, float* __restrict__ out)
{
    extern __shared__ char smem[];
    __bf16* xs  = (__bf16*)smem;                 // [MT][XSS]
    __bf16* hs  = xs + MT * XSS;                 // [MT][HSS]
    int*    tks = (int*)(hs + MT * HSS);         // [MT]
    float*  pv  = (float*)(tks + MT);            // [MT]

    const int e  = blockIdx.y;
    const int ne = counts[e];
    const int t0 = blockIdx.x * MT;
    if (t0 >= ne) return;

    const int tid  = threadIdx.x;
    const int wave = tid >> 5;        // 0..15
    const int lane = tid & 31;
    const int ln   = lane & 15;
    const int kh   = lane >> 4;

    if (tid < MT) {
        int idx = t0 + tid;
        int tk = (idx < ne) ? perm[e * T_TOK + idx] : -1;
        tks[tid] = tk;
        pv[tid]  = (tk >= 0) ? prob[tk] : 0.f;
    }
    __syncthreads();

    // stage x tile (f32 -> bf16): 16 threads per row, 64 cols each
    {
        int r  = tid >> 4;            // 0..31
        int c0 = (tid & 15) * 64;
        int tk = tks[r];
        __bf16* dst = xs + r * XSS + c0;
        if (tk >= 0) {
            const float4* src = (const float4*)(x + (size_t)tk * DIM + c0);
#pragma unroll
            for (int i = 0; i < 16; ++i) {
                float4 v = src[i];
                dst[4 * i + 0] = (__bf16)v.x; dst[4 * i + 1] = (__bf16)v.y;
                dst[4 * i + 2] = (__bf16)v.z; dst[4 * i + 3] = (__bf16)v.w;
            }
        } else {
#pragma unroll
            for (int i = 0; i < 64; ++i) dst[i] = (__bf16)0.f;
        }
    }
    __syncthreads();

    const __bf16* xrL = xs + ln * XSS;
    const __bf16* xrH = xs + (16 + ln) * XSS;

    // ---- GEMM1: h = relu(x @ W1[e] + b1[e]) ----
    // 128 n-tiles; 16 waves x 4 groups x 2 tiles, each tile x 2 row halves.
    const __bf16* W1b = W1t + (size_t)e * HID * DIM + 16 * kh;
#pragma unroll 1
    for (int g = 0; g < 4; ++g) {
        const int nt0 = (g * 16 + wave) * 2;
        const int h0  = nt0 * 16;
        const __bf16* bc0 = W1b + (size_t)(h0 + ln) * DIM;
        const __bf16* bc1 = bc0 + (size_t)16 * DIM;

        v8f aL0 = {}, aL1 = {}, aH0 = {}, aH1 = {};
        v16bf aL = ldA(xrL, 0, kh), aH = ldA(xrH, 0, kh);
        v16bf b0 = ldB(bc0, 0),     b1v = ldB(bc1, 0);
#pragma unroll 2
        for (int kk = 0; kk < DIM - 32; kk += 32) {
            v16bf aLn = ldA(xrL, kk + 32, kh);
            v16bf aHn = ldA(xrH, kk + 32, kh);
            v16bf b0n = ldB(bc0, kk + 32);
            v16bf b1n = ldB(bc1, kk + 32);
            aL0 = wmma_bf16(aL, b0,  aL0);
            aL1 = wmma_bf16(aL, b1v, aL1);
            aH0 = wmma_bf16(aH, b0,  aH0);
            aH1 = wmma_bf16(aH, b1v, aH1);
            aL = aLn; aH = aHn; b0 = b0n; b1v = b1n;
        }
        aL0 = wmma_bf16(aL, b0,  aL0);
        aL1 = wmma_bf16(aL, b1v, aL1);
        aH0 = wmma_bf16(aH, b0,  aH0);
        aH1 = wmma_bf16(aH, b1v, aH1);

        float bi0 = b1[e * HID + h0 + ln];
        float bi1 = b1[e * HID + h0 + 16 + ln];
#pragma unroll
        for (int r = 0; r < 8; ++r) {
            int mL = r + 8 * kh, mH = 16 + mL;
            float v;
            v = aL0[r] + bi0; v = v > 0.f ? v : 0.f; hs[mL * HSS + h0 + ln] = (__bf16)v;
            v = aL1[r] + bi1; v = v > 0.f ? v : 0.f; hs[mL * HSS + h0 + 16 + ln] = (__bf16)v;
            v = aH0[r] + bi0; v = v > 0.f ? v : 0.f; hs[mH * HSS + h0 + ln] = (__bf16)v;
            v = aH1[r] + bi1; v = v > 0.f ? v : 0.f; hs[mH * HSS + h0 + 16 + ln] = (__bf16)v;
        }
    }
    __syncthreads();

    const __bf16* hrL = hs + ln * HSS;
    const __bf16* hrH = hs + (16 + ln) * HSS;

    // ---- GEMM2: y = h @ W2[e] + b2[e]; scale by prob; scatter rows ----
    // 64 n-tiles; 16 waves x 2 groups x 2 tiles x 2 row halves.
    const __bf16* W2b = W2t + (size_t)e * DIM * HID + 16 * kh;
#pragma unroll 1
    for (int g = 0; g < 2; ++g) {
        const int nt0 = (g * 16 + wave) * 2;
        const int d0  = nt0 * 16;
        const __bf16* bc0 = W2b + (size_t)(d0 + ln) * HID;
        const __bf16* bc1 = bc0 + (size_t)16 * HID;

        v8f aL0 = {}, aL1 = {}, aH0 = {}, aH1 = {};
        v16bf aL = ldA(hrL, 0, kh), aH = ldA(hrH, 0, kh);
        v16bf b0 = ldB(bc0, 0),     b1v = ldB(bc1, 0);
#pragma unroll 2
        for (int kk = 0; kk < HID - 32; kk += 32) {
            v16bf aLn = ldA(hrL, kk + 32, kh);
            v16bf aHn = ldA(hrH, kk + 32, kh);
            v16bf b0n = ldB(bc0, kk + 32);
            v16bf b1n = ldB(bc1, kk + 32);
            aL0 = wmma_bf16(aL, b0,  aL0);
            aL1 = wmma_bf16(aL, b1v, aL1);
            aH0 = wmma_bf16(aH, b0,  aH0);
            aH1 = wmma_bf16(aH, b1v, aH1);
            aL = aLn; aH = aHn; b0 = b0n; b1v = b1n;
        }
        aL0 = wmma_bf16(aL, b0,  aL0);
        aL1 = wmma_bf16(aL, b1v, aL1);
        aH0 = wmma_bf16(aH, b0,  aH0);
        aH1 = wmma_bf16(aH, b1v, aH1);

        float bi0 = b2[e * DIM + d0 + ln];
        float bi1 = b2[e * DIM + d0 + 16 + ln];
#pragma unroll
        for (int r = 0; r < 8; ++r) {
            int mL = r + 8 * kh, mH = 16 + mL;
            int tkL = tks[mL], tkH = tks[mH];
            if (tkL >= 0) {
                float* o = out + (size_t)tkL * DIM + d0 + ln;
                o[0]  = (aL0[r] + bi0) * pv[mL];
                o[16] = (aL1[r] + bi1) * pv[mL];
            }
            if (tkH >= 0) {
                float* o = out + (size_t)tkH * DIM + d0 + ln;
                o[0]  = (aH0[r] + bi0) * pv[mH];
                o[16] = (aH1[r] + bi1) * pv[mH];
            }
        }
    }
}

extern "C" void kernel_launch(void* const* d_in, const int* in_sizes, int n_in,
                              void* d_out, int out_size, void* d_ws, size_t ws_size,
                              hipStream_t stream)
{
    const float* x  = (const float*)d_in[0];
    const float* Wr = (const float*)d_in[1];
    const float* br = (const float*)d_in[2];
    const float* W1 = (const float*)d_in[3];
    const float* b1 = (const float*)d_in[4];
    const float* W2 = (const float*)d_in[5];
    const float* b2 = (const float*)d_in[6];
    float* out = (float*)d_out;

    char* ws = (char*)d_ws;
    int*    expert = (int*)(ws + OFF_EXPERT);
    float*  prob   = (float*)(ws + OFF_PROB);
    int*    counts = (int*)(ws + OFF_COUNTS);
    int*    perm   = (int*)(ws + OFF_PERM);
    __bf16* W1t    = (__bf16*)(ws + OFF_W1T);
    __bf16* W2t    = (__bf16*)(ws + OFF_W2T);

    k_router<<<T_TOK, 64, 0, stream>>>(x, Wr, br, expert, prob, counts);
    k_scatter<<<T_TOK / 256, 256, 0, stream>>>(expert, counts, perm);
    k_tcvt<<<dim3(HID / 32, DIM / 32, NEXP), 256, 0, stream>>>(W1, W1t, DIM, HID);
    k_tcvt<<<dim3(DIM / 32, HID / 32, NEXP), 256, 0, stream>>>(W2, W2t, HID, DIM);

    size_t smem = (size_t)MT * XSS * 2 + (size_t)MT * HSS * 2 + MT * 8;
    // = 66048 + 131584 + 256 = 197888 B of the WGP's 320 KB LDS
    k_moe<<<dim3(T_TOK / MT, NEXP), 512, smem, stream>>>(
        x, W1t, b1, W2t, b2, perm, counts, prob, out);
}
